// AttnDecoderRNN_46024869543986
// MI455X (gfx1250) — matrix operand order repaired and checked by reading, not compiled, not run on hardware
//
#include <hip/hip_runtime.h>
#include <hip/hip_bf16.h>

// ---------------- problem dims ----------------
constexpr int kH  = 64;    // hidden
constexpr int kB  = 512;   // batch
constexpr int kT  = 512;   // time
constexpr int kEH = 128;   // encoder feature dim (2H)

// ---------------- WMMA types ----------------
typedef __attribute__((ext_vector_type(16))) _Float16 v16h;
typedef __attribute__((ext_vector_type(8)))  float    v8f;

// ---------------- workspace layout (byte offsets) ----------------
// xcat_h : f16 [512][256]      @ 0        (262144 B)
// h0_h   : f16 [2][512][64]    @ 262144   (131072 B)
// x_h    : f16 [512][64]       @ 393216   ( 65536 B)
// gates  : f32 [2][512][256]   @ 458752   (1048576 B)
constexpr size_t WS_XCAT  = 0;
constexpr size_t WS_H0H   = 262144;
constexpr size_t WS_XH    = 393216;
constexpr size_t WS_GATES = 458752;

__device__ __forceinline__ float wave_reduce_sum(float v) {
#pragma unroll
  for (int off = 16; off >= 1; off >>= 1) v += __shfl_xor(v, off, 32);
  return v;
}

__device__ __forceinline__ void wait_asynccnt0() {
#if __has_builtin(__builtin_amdgcn_s_wait_asynccnt)
  __builtin_amdgcn_s_wait_asynccnt(0);
#else
  asm volatile("s_wait_asynccnt 0x0" ::: "memory");
#endif
}

// ---- A fragment: 16x32 f16, row-major source (stride lda halves), tile at (row0, k0) ----
// lanes 0-15 -> M=lane, K=k0..k0+15 ; lanes 16-31 -> K=k0+16..k0+31
__device__ __forceinline__ v16h load_a_f16(const _Float16* __restrict__ A, int lda,
                                           int row0, int k0, int lane) {
  const int m  = row0 + (lane & 15);
  const int kb = k0 + ((lane >> 4) << 4);        // 0 or 16
  const _Float16* p = A + (size_t)m * lda + kb;
  v16h a;
#pragma unroll
  for (int j = 0; j < 16; ++j) a[j] = p[j];
  return a;
}

// ---- B fragment: 32x16, where B[k][n] = W[n][k] with W row-major (N x K) f32, stride ldw ----
// lanes 0-15 -> N=lane, K=k0..k0+15 ; lanes 16-31 -> K=k0+16..k0+31
__device__ __forceinline__ v16h load_b_wT_f32(const float* __restrict__ W, int ldw,
                                              int k0, int col0, int lane) {
  const int n  = col0 + (lane & 15);
  const int kb = k0 + ((lane >> 4) << 4);        // 0 or 16
  const float* p = W + (size_t)n * ldw + kb;
  v16h b;
#pragma unroll
  for (int j = 0; j < 16; ++j) b[j] = (_Float16)p[j];
  return b;
}

// =====================================================================
// Kernel 1: fused attention (scores + softmax + weighted sum) per batch row.
// One workgroup per b; 256 threads (8 waves). The 256 KB row enc[b,:,:] is
// streamed into LDS ONCE via CDNA5 async-to-LDS DMA (no VGPR round-trip),
// then both the score pass and the weighted-sum pass read from LDS.
// =====================================================================
__global__ void attn_softmax_kernel(const float* __restrict__ h0,
                                    const float* __restrict__ enc,
                                    const float* __restrict__ W_attn,
                                    const float* __restrict__ b_attn,
                                    const int*  __restrict__ di_ptr,
                                    _Float16*   __restrict__ xcat_h) {
  extern __shared__ float smem[];
  float* s_enc    = smem;                       // [512*128]  (256 KB)
  float* s_scores = s_enc + kT * kEH;           // [512]
  float* s_red    = s_scores + kT;              // [256]

  const int tid  = threadIdx.x;
  const int lane = tid & 31;
  const int wave = tid >> 5;
  const int b    = blockIdx.x;

  // ---- kick off async copy: enc[b,:,:] (256 KB) -> LDS ----
  // 16384 x 16B chunks; 256 threads x 64 iterations; one async b128 per
  // wave-iteration moves 512B directly into LDS (tracked by ASYNCcnt).
  {
    unsigned long long ga =
        (unsigned long long)(uintptr_t)(enc + (size_t)b * kT * kEH) +
        (unsigned long long)tid * 16ull;
    unsigned int la = (unsigned int)(uintptr_t)s_enc + (unsigned int)tid * 16u;
#pragma unroll 4
    for (int i = 0; i < 64; ++i) {
      asm volatile("global_load_async_to_lds_b128 %0, %1, off"
                   :: "v"(la), "v"(ga) : "memory");
      ga += 4096ull;   // 256 threads * 16B
      la += 4096u;
    }
  }

  // ---- const_b = dec_hidden[b,:] . W_attn[128:256] + b_attn (overlaps DMA) ----
  // dec_hidden row b == flat h0 elements [b*128, b*128+128)
  float p = 0.f;
  if (tid < 128) p = h0[(size_t)b * 128 + tid] * W_attn[128 + tid];
  s_red[tid] = p;
  __syncthreads();
  for (int s = 128; s > 0; s >>= 1) {
    if (tid < s) s_red[tid] += s_red[tid + s];
    __syncthreads();
  }
  const float const_b = s_red[0] + b_attn[0];
  const float4 wa = reinterpret_cast<const float4*>(W_attn)[lane];  // W_attn[lane*4..+3]

  // ---- drain DMA, then everyone sees the staged row ----
  wait_asynccnt0();
  __syncthreads();

  // ---- pass 1: scores from LDS ----
  const float4* s_enc4 = reinterpret_cast<const float4*>(s_enc);
  for (int t = wave; t < kT; t += 8) {
    float4 v = s_enc4[t * 32 + lane];
    float d = v.x * wa.x + v.y * wa.y + v.z * wa.z + v.w * wa.w;
    d = wave_reduce_sum(d);
    if (lane == 0) s_scores[t] = d + const_b;
  }
  __syncthreads();

  // ---- softmax over t (512 entries, 2 per thread) ----
  float a0 = s_scores[tid], a1 = s_scores[tid + 256];
  s_red[tid] = fmaxf(a0, a1);
  __syncthreads();
  for (int s = 128; s > 0; s >>= 1) {
    if (tid < s) s_red[tid] = fmaxf(s_red[tid], s_red[tid + s]);
    __syncthreads();
  }
  const float smax = s_red[0];
  __syncthreads();
  const float e0 = expf(a0 - smax), e1 = expf(a1 - smax);
  s_red[tid] = e0 + e1;
  __syncthreads();
  for (int s = 128; s > 0; s >>= 1) {
    if (tid < s) s_red[tid] += s_red[tid + s];
    __syncthreads();
  }
  const float inv = 1.0f / s_red[0];
  __syncthreads();
  s_scores[tid]       = e0 * inv;
  s_scores[tid + 256] = e1 * inv;
  __syncthreads();

  // ---- pass 2: attn_applied[h] = sum_t w[t] * enc[b,t,h]  (from LDS) ----
  const int ch = tid & 127;            // channel
  const int t0 = (tid >> 7) * 256;     // half of the time range
  float acc = 0.f;
#pragma unroll 4
  for (int t = 0; t < 256; ++t)
    acc += s_scores[t0 + t] * s_enc[(size_t)(t0 + t) * kEH + ch];
  s_red[tid] = acc;
  __syncthreads();
  if (tid < 128) {
    const float tot = s_red[tid] + s_red[tid + 128];
    xcat_h[(size_t)b * 256 + 128 + tid] = (_Float16)tot;             // attn part
    const int di = di_ptr[0];
    xcat_h[(size_t)b * 256 + tid] =
        (_Float16)enc[((size_t)b * kT + di) * kEH + tid];            // enc[:,di,:] part
  }
}

// =====================================================================
// Kernel 2: f32 -> f16 convert (for h0 used as WMMA A operand)
// =====================================================================
__global__ void cvt_f32_to_f16_kernel(const float* __restrict__ src,
                                      _Float16* __restrict__ dst, int n) {
  int i = blockIdx.x * blockDim.x + threadIdx.x;
  if (i < n) dst[i] = (_Float16)src[i];
}

// =====================================================================
// Kernel 3: combine GEMM  x = relu(xcat (512x256) @ W_comb^T (256x64) + b)
// 4 waves/block, each wave owns one 16x16 N-tile; block owns one M-tile row.
// =====================================================================
__global__ void combine_gemm_kernel(const _Float16* __restrict__ xcat_h,
                                    const float* __restrict__ W_comb,
                                    const float* __restrict__ b_comb,
                                    _Float16* __restrict__ x_h) {
  const int lane = threadIdx.x & 31;
  const int wv   = threadIdx.x >> 5;       // 0..3 -> N tile
  const int m0   = blockIdx.x * 16;
  const int n0   = wv * 16;

  v8f c = {};
#pragma unroll
  for (int kt = 0; kt < 8; ++kt) {
    const int k0 = kt * 32;
    v16h a  = load_a_f16(xcat_h, 256, m0, k0, lane);
    v16h bq = load_b_wT_f32(W_comb, 256, k0, n0, lane);
    c = __builtin_amdgcn_wmma_f32_16x16x32_f16(false, a, false, bq,
                                               (short)0, c, false, false);
  }
  const int n    = n0 + (lane & 15);
  const float bs = b_comb[n];
  const int mrow = m0 + ((lane >> 4) << 3);
#pragma unroll
  for (int r = 0; r < 8; ++r) {
    float v = fmaxf(c[r] + bs, 0.f);
    x_h[(size_t)(mrow + r) * kH + n] = (_Float16)v;
  }
}

// =====================================================================
// Kernel 4: LSTM gate GEMMs (both directions via blockIdx.z)
// gates[d] = x @ w_ih_d^T + h0[d] @ w_hh_d^T + (b_ih_d + b_hh_d)
// =====================================================================
__global__ void gates_gemm_kernel(const _Float16* __restrict__ x_h,
                                  const _Float16* __restrict__ h0_h,
                                  const float* __restrict__ w_ih_f, const float* __restrict__ w_hh_f,
                                  const float* __restrict__ b_ih_f, const float* __restrict__ b_hh_f,
                                  const float* __restrict__ w_ih_r, const float* __restrict__ w_hh_r,
                                  const float* __restrict__ b_ih_r, const float* __restrict__ b_hh_r,
                                  float* __restrict__ gates) {
  const int lane = threadIdx.x & 31;
  const int wv   = threadIdx.x >> 5;                 // 0..3
  const int m0   = blockIdx.x * 16;
  const int n0   = (blockIdx.y * 4 + wv) * 16;       // N = 256 -> 16 tiles
  const int d    = blockIdx.z;

  const float* wih = d ? w_ih_r : w_ih_f;
  const float* whh = d ? w_hh_r : w_hh_f;
  const float* bih = d ? b_ih_r : b_ih_f;
  const float* bhh = d ? b_hh_r : b_hh_f;
  const _Float16* hA = h0_h + (size_t)d * kB * kH;

  v8f c = {};
#pragma unroll
  for (int kt = 0; kt < 2; ++kt) {                   // K = 64, x @ w_ih^T
    v16h a  = load_a_f16(x_h, kH, m0, kt * 32, lane);
    v16h bq = load_b_wT_f32(wih, kH, kt * 32, n0, lane);
    c = __builtin_amdgcn_wmma_f32_16x16x32_f16(false, a, false, bq,
                                               (short)0, c, false, false);
  }
#pragma unroll
  for (int kt = 0; kt < 2; ++kt) {                   // + h0[d] @ w_hh^T
    v16h a  = load_a_f16(hA, kH, m0, kt * 32, lane);
    v16h bq = load_b_wT_f32(whh, kH, kt * 32, n0, lane);
    c = __builtin_amdgcn_wmma_f32_16x16x32_f16(false, a, false, bq,
                                               (short)0, c, false, false);
  }
  const int n    = n0 + (lane & 15);
  const float bs = bih[n] + bhh[n];
  float* gd = gates + (size_t)d * kB * 256;
  const int mrow = m0 + ((lane >> 4) << 3);
#pragma unroll
  for (int r = 0; r < 8; ++r)
    gd[(size_t)(mrow + r) * 256 + n] = c[r] + bs;
}

// =====================================================================
// Kernel 5: elementwise LSTM cell + output assembly
// d_out = [output (512x128)] [h_new (2x512x64)] [c_new (2x512x64)]
// =====================================================================
__global__ void lstm_elem_kernel(const float* __restrict__ gates,
                                 const float* __restrict__ c0,
                                 float* __restrict__ out) {
  const int i = blockIdx.x * blockDim.x + threadIdx.x;   // d*32768 + b*64 + hh
  if (i >= 2 * kB * kH) return;
  const int hh = i & 63;
  const int b  = (i >> 6) & 511;
  const int d  = i >> 15;
  const float* g = gates + ((size_t)d * kB + b) * 256;
  const float gi = g[hh], gf = g[64 + hh], gg = g[128 + hh], go = g[192 + hh];
  const float cv = c0[i];
  const float si = 1.f / (1.f + expf(-gi));
  const float sf = 1.f / (1.f + expf(-gf));
  const float so = 1.f / (1.f + expf(-go));
  const float cn = sf * cv + si * tanhf(gg);
  const float hn = so * tanhf(cn);
  out[(size_t)b * 128 + d * kH + hh] = hn;   // output (b,1,128)
  out[65536 + i]  = hn;                      // h_new
  out[131072 + i] = cn;                      // c_new
}

// =====================================================================
extern "C" void kernel_launch(void* const* d_in, const int* in_sizes, int n_in,
                              void* d_out, int out_size, void* d_ws, size_t ws_size,
                              hipStream_t stream) {
  (void)in_sizes; (void)n_in; (void)out_size; (void)ws_size;
  const float* h0     = (const float*)d_in[0];
  const float* c0     = (const float*)d_in[1];
  const float* enc    = (const float*)d_in[2];
  const float* W_attn = (const float*)d_in[3];
  const float* b_attn = (const float*)d_in[4];
  const float* W_comb = (const float*)d_in[5];
  const float* b_comb = (const float*)d_in[6];
  const float* w_ih_f = (const float*)d_in[7];
  const float* w_hh_f = (const float*)d_in[8];
  const float* b_ih_f = (const float*)d_in[9];
  const float* b_hh_f = (const float*)d_in[10];
  const float* w_ih_r = (const float*)d_in[11];
  const float* w_hh_r = (const float*)d_in[12];
  const float* b_ih_r = (const float*)d_in[13];
  const float* b_hh_r = (const float*)d_in[14];
  const int*   di     = (const int*)d_in[15];
  float* out = (float*)d_out;

  char* ws = (char*)d_ws;
  _Float16* xcat_h = (_Float16*)(ws + WS_XCAT);
  _Float16* h0_h   = (_Float16*)(ws + WS_H0H);
  _Float16* x_h    = (_Float16*)(ws + WS_XH);
  float*    gates  = (float*)(ws + WS_GATES);

  // 256 KB enc-row stage + scores + reduction scratch (CDNA5: 320 KB LDS/WGP)
  const size_t attn_lds = (size_t)(kT * kEH + kT + 256) * sizeof(float);
  hipFuncSetAttribute((const void*)attn_softmax_kernel,
                      hipFuncAttributeMaxDynamicSharedMemorySize, (int)attn_lds);

  attn_softmax_kernel<<<kB, 256, attn_lds, stream>>>(h0, enc, W_attn, b_attn, di, xcat_h);
  cvt_f32_to_f16_kernel<<<(2 * kB * kH + 255) / 256, 256, 0, stream>>>(h0, h0_h, 2 * kB * kH);
  combine_gemm_kernel<<<kB / 16, 128, 0, stream>>>(xcat_h, W_comb, b_comb, x_h);
  gates_gemm_kernel<<<dim3(kB / 16, 4, 2), 128, 0, stream>>>(
      x_h, h0_h, w_ih_f, w_hh_f, b_ih_f, b_hh_f,
      w_ih_r, w_hh_r, b_ih_r, b_hh_r, gates);
  lstm_elem_kernel<<<(2 * kB * kH + 255) / 256, 256, 0, stream>>>(gates, c0, out);
}
